// LSTMCellDRCNet_33380485825237
// MI455X (gfx1250) — compile-verified, matrix operand order checked by reading, not run on hardware
//
#include <hip/hip_runtime.h>
#include <hip/hip_bf16.h>

// Problem sizes (reference): T=128 but only hs[13], hs[35] are used -> 36 steps.
#define TB      512          // batch
#define II      512          // input size
#define HH      1024         // hidden
#define GG      4096         // 4*H
#define KK      1536         // I + H (concat GEMM K)
#define TSTEPS  36
#define P1      13
#define P2      35
#define KITERS  (KK / 32)    // 48

#define LDS_ROW_B   80       // 64B of data + 16B TDM pad per row (bank-conflict-free)

typedef __attribute__((ext_vector_type(16))) __bf16           bf16x16;
typedef __attribute__((ext_vector_type(8)))  float            f32x8;
typedef __attribute__((ext_vector_type(8)))  unsigned short   u16x8;
typedef __attribute__((ext_vector_type(4)))  unsigned int     u32x4;
typedef __attribute__((ext_vector_type(4)))  int              i32x4;
typedef __attribute__((ext_vector_type(8)))  int              i32x8;

union FragU { bf16x16 v; u16x8 h[2]; };

__device__ __forceinline__ unsigned short f32_to_bf16_rne(float f) {
    unsigned int u = __float_as_uint(f);
    u += 0x7FFFu + ((u >> 16) & 1u);       // round-to-nearest-even
    return (unsigned short)(u >> 16);
}

__device__ __forceinline__ float sigmoidf_fast(float x) {
    return 1.0f / (1.0f + __expf(-x));
}

// -------------------------------------------------------------------------
// One-time conversion kernels
// -------------------------------------------------------------------------
__global__ void build_wcat_kernel(const float* __restrict__ w_ih,
                                  const float* __restrict__ w_hh,
                                  unsigned short* __restrict__ wcat) {
    int idx = blockIdx.x * blockDim.x + threadIdx.x;   // 4096*1536
    int n = idx / KK;
    int k = idx % KK;
    float v = (k < II) ? w_ih[n * II + k] : w_hh[n * HH + (k - II)];
    wcat[idx] = f32_to_bf16_rne(v);
}

__global__ void convert_x_kernel(const float* __restrict__ x,
                                 unsigned short* __restrict__ xbf) {
    int idx = blockIdx.x * blockDim.x + threadIdx.x;   // 36*512*512
    xbf[idx] = f32_to_bf16_rne(x[idx]);
}

__global__ void init_state_kernel(unsigned short* __restrict__ hbf,
                                  float* __restrict__ c) {
    int idx = blockIdx.x * blockDim.x + threadIdx.x;   // 512*1024
    hbf[idx] = 0;
    c[idx]   = 0.0f;
}

// -------------------------------------------------------------------------
// Tensor Data Mover: 2D tile (rows x 32 bf16) global -> LDS, padded rows.
// D# built per CDNA5 ISA ch.8: group0 = count/lds_addr/global_addr/type,
// group1 = data_size, pad (16 DW interval / 4 DW amount), dims, stride.
// -------------------------------------------------------------------------
#if __has_builtin(__builtin_amdgcn_tensor_load_to_lds)
#define HAVE_TDM 1
__device__ __forceinline__ void tdm_load_tile(unsigned lds_off,
                                              const unsigned short* gptr,
                                              unsigned rows,
                                              unsigned stride_elems) {
    unsigned long long ga = (unsigned long long)(size_t)gptr;
    u32x4 g0;
    g0[0] = 1u;                                          // count=1 (valid user D#)
    g0[1] = lds_off;                                     // lds_addr (bytes)
    g0[2] = (unsigned)(ga & 0xFFFFFFFFu);                // global_addr[31:0]
    g0[3] = (unsigned)((ga >> 32) & 0x01FFFFFFu)         // global_addr[56:32]
          | (2u << 30);                                  // type=2 ("image")
    i32x8 g1;
    g1[0] = (1 << 16)                                    // data_size=1 (2 bytes)
          | (1 << 20)                                    // pad_enable
          | (3 << 22)                                    // pad_interval: 16 DWORDs (64B row)
          | (3 << 25);                                   // pad_amount: 4 DWORDs (16B)
    g1[1] = (int)(32u << 16);                            // tensor_dim0 = 32 (lo 16 in [63:48])
    g1[2] = (int)((rows & 0xFFFFu) << 16);               // tensor_dim0 hi=0 | tensor_dim1 lo
    g1[3] = (int)(32u << 16);                            // tensor_dim1 hi=0 | tile_dim0 = 32
    g1[4] = (int)(rows & 0xFFFFu);                       // tile_dim1 = rows | tile_dim2 = 0
    g1[5] = (int)stride_elems;                           // tensor_dim0_stride lo
    g1[6] = 0;                                           // stride hi | dim1_stride lo
    g1[7] = 0;
    i32x4 gz = {0, 0, 0, 0};                             // 2D tile: groups 2/3 unused
#if defined(__clang_major__) && __clang_major__ >= 23
    __builtin_amdgcn_tensor_load_to_lds(g0, g1, gz, gz, (i32x8)0, 0);
#else
    __builtin_amdgcn_tensor_load_to_lds(g0, g1, gz, gz, 0);
#endif
}
#else
#define HAVE_TDM 0
#endif

// -------------------------------------------------------------------------
// Fragment fetch from a padded LDS tile (rows x 32 bf16, row stride 80B).
// WMMA bf16 layouts (ISA 7.12.2): lane L<16 -> row/col = L, K {0..7,16..23};
// lane>=16 -> same row/col, K {8..15,24..31}.
// -------------------------------------------------------------------------
__device__ __forceinline__ bf16x16 lds_frag(const unsigned char* buf, int row0, int lane) {
    const unsigned char* p = buf + (row0 + (lane & 15)) * LDS_ROW_B
                                 + ((lane >> 4) << 4);   // +0 or +16 bytes (K half)
    FragU f;
    f.h[0] = *(const u16x8*)(p);
    f.h[1] = *(const u16x8*)(p + 32);                    // +16 elements
    return f.v;
}

// -------------------------------------------------------------------------
// Recurrent step GEMM: gates[512,4096] = [x_t | h] @ Wcat^T (bf16 WMMA, f32 acc)
// WG: 8 waves (2x4) -> 64(M) x 256(N); wave: 32x64; grid (16,8).
// TDM double-buffers A (64x32) and B (256x32) tiles in LDS.
// -------------------------------------------------------------------------
__global__ void __launch_bounds__(256)
lstm_gemm_kernel(const unsigned short* __restrict__ xbf,   // [36,512,512]
                 const unsigned short* __restrict__ hbf,   // [512,1024]
                 const unsigned short* __restrict__ wcat,  // [4096,1536]
                 float* __restrict__ gates,                // [512,4096]
                 int t) {
    __shared__ __align__(16) unsigned char ldsA[2][64  * LDS_ROW_B];   // 10 KB
    __shared__ __align__(16) unsigned char ldsB[2][256 * LDS_ROW_B];   // 40 KB

    const int lane = threadIdx.x & 31;
    const int wave = threadIdx.x >> 5;
    const int wm = wave >> 2;                 // 0..1 (M)
    const int wn = wave & 3;                  // 0..3 (N)
    const int mwg = blockIdx.y * 64;          // WG M origin
    const int nwg = blockIdx.x * 256;         // WG N origin

    const unsigned short* xt = xbf + (size_t)t * TB * II;

    f32x8 acc[2][4];
#pragma unroll
    for (int i = 0; i < 2; ++i)
#pragma unroll
        for (int j = 0; j < 4; ++j)
            acc[i][j] = (f32x8)0.0f;

#if HAVE_TDM
    // ---- TDM double-buffered pipeline -----------------------------------
    auto issue_tiles = [&](int buf, int it) {
        int k0 = it * 32;
        const unsigned short* abase; int lda, ka;
        if (k0 < II) { abase = xt;  lda = II; ka = k0; }
        else         { abase = hbf; lda = HH; ka = k0 - II; }
        tdm_load_tile((unsigned)(size_t)&ldsA[buf][0],
                      abase + (size_t)mwg * lda + ka, 64, (unsigned)lda);
        tdm_load_tile((unsigned)(size_t)&ldsB[buf][0],
                      wcat + (size_t)nwg * KK + k0, 256, KK);
    };

    if (wave == 0) issue_tiles(0, 0);

    for (int it = 0; it < KITERS; ++it) {
        const int cur = it & 1;
        if (wave == 0) __builtin_amdgcn_s_wait_tensorcnt(0);
        __syncthreads();                               // buf[cur] ready for all
        if (wave == 0 && it + 1 < KITERS) issue_tiles(cur ^ 1, it + 1);

        bf16x16 a0 = lds_frag(ldsA[cur], wm * 32 +  0, lane);
        bf16x16 a1 = lds_frag(ldsA[cur], wm * 32 + 16, lane);
        bf16x16 b0 = lds_frag(ldsB[cur], wn * 64 +  0, lane);
        bf16x16 b1 = lds_frag(ldsB[cur], wn * 64 + 16, lane);
        bf16x16 b2 = lds_frag(ldsB[cur], wn * 64 + 32, lane);
        bf16x16 b3 = lds_frag(ldsB[cur], wn * 64 + 48, lane);

        acc[0][0] = __builtin_amdgcn_wmma_f32_16x16x32_bf16(false, a0, false, b0, (short)0, acc[0][0], false, false);
        acc[0][1] = __builtin_amdgcn_wmma_f32_16x16x32_bf16(false, a0, false, b1, (short)0, acc[0][1], false, false);
        acc[0][2] = __builtin_amdgcn_wmma_f32_16x16x32_bf16(false, a0, false, b2, (short)0, acc[0][2], false, false);
        acc[0][3] = __builtin_amdgcn_wmma_f32_16x16x32_bf16(false, a0, false, b3, (short)0, acc[0][3], false, false);
        acc[1][0] = __builtin_amdgcn_wmma_f32_16x16x32_bf16(false, a1, false, b0, (short)0, acc[1][0], false, false);
        acc[1][1] = __builtin_amdgcn_wmma_f32_16x16x32_bf16(false, a1, false, b1, (short)0, acc[1][1], false, false);
        acc[1][2] = __builtin_amdgcn_wmma_f32_16x16x32_bf16(false, a1, false, b2, (short)0, acc[1][2], false, false);
        acc[1][3] = __builtin_amdgcn_wmma_f32_16x16x32_bf16(false, a1, false, b3, (short)0, acc[1][3], false, false);

        __syncthreads();                               // reads of buf[cur] done
    }
#else
    // ---- Fallback: cooperative staging, single buffer -------------------
    const int tid = (int)threadIdx.x;
    for (int it = 0; it < KITERS; ++it) {
        int k0 = it * 32;
        const unsigned short* abase; int lda, ka;
        if (k0 < II) { abase = xt;  lda = II; ka = k0; }
        else         { abase = hbf; lda = HH; ka = k0 - II; }
        __syncthreads();
        {   // A: 64 rows * 4 vec8 = 256 chunks (1/thread)
            int row = tid >> 2, v = tid & 3;
            *(u16x8*)&ldsA[0][row * LDS_ROW_B + v * 16] =
                *((const u16x8*)(abase + (size_t)(mwg + row) * lda + ka) + v);
        }
#pragma unroll
        for (int q = 0; q < 4; ++q) {   // B: 256 rows * 4 vec8 = 1024 chunks
            int chunk = tid + 256 * q;
            int row = chunk >> 2, v = chunk & 3;
            *(u16x8*)&ldsB[0][row * LDS_ROW_B + v * 16] =
                *((const u16x8*)(wcat + (size_t)(nwg + row) * KK + k0) + v);
        }
        __syncthreads();

        bf16x16 a0 = lds_frag(ldsA[0], wm * 32 +  0, lane);
        bf16x16 a1 = lds_frag(ldsA[0], wm * 32 + 16, lane);
        bf16x16 b0 = lds_frag(ldsB[0], wn * 64 +  0, lane);
        bf16x16 b1 = lds_frag(ldsB[0], wn * 64 + 16, lane);
        bf16x16 b2 = lds_frag(ldsB[0], wn * 64 + 32, lane);
        bf16x16 b3 = lds_frag(ldsB[0], wn * 64 + 48, lane);

        acc[0][0] = __builtin_amdgcn_wmma_f32_16x16x32_bf16(false, a0, false, b0, (short)0, acc[0][0], false, false);
        acc[0][1] = __builtin_amdgcn_wmma_f32_16x16x32_bf16(false, a0, false, b1, (short)0, acc[0][1], false, false);
        acc[0][2] = __builtin_amdgcn_wmma_f32_16x16x32_bf16(false, a0, false, b2, (short)0, acc[0][2], false, false);
        acc[0][3] = __builtin_amdgcn_wmma_f32_16x16x32_bf16(false, a0, false, b3, (short)0, acc[0][3], false, false);
        acc[1][0] = __builtin_amdgcn_wmma_f32_16x16x32_bf16(false, a1, false, b0, (short)0, acc[1][0], false, false);
        acc[1][1] = __builtin_amdgcn_wmma_f32_16x16x32_bf16(false, a1, false, b1, (short)0, acc[1][1], false, false);
        acc[1][2] = __builtin_amdgcn_wmma_f32_16x16x32_bf16(false, a1, false, b2, (short)0, acc[1][2], false, false);
        acc[1][3] = __builtin_amdgcn_wmma_f32_16x16x32_bf16(false, a1, false, b3, (short)0, acc[1][3], false, false);
    }
#endif

    // Store: C/D layout — VGPR r: lanes 0..15 -> M=r, lanes 16..31 -> M=r+8; N = lane&15
    const int col_in_tile = lane & 15;
    const int row_half    = (lane >> 4) * 8;
#pragma unroll
    for (int i = 0; i < 2; ++i) {
#pragma unroll
        for (int j = 0; j < 4; ++j) {
            int n = nwg + wn * 64 + j * 16 + col_in_tile;
#pragma unroll
            for (int r = 0; r < 8; ++r) {
                int m = mwg + wm * 32 + i * 16 + row_half + r;
                gates[(size_t)m * GG + n] = acc[i][j][r];
            }
        }
    }
}

// -------------------------------------------------------------------------
// LSTM cell elementwise update (gate order i,f,g,o along 4H dim)
// -------------------------------------------------------------------------
__global__ void lstm_cell_kernel(const float* __restrict__ gates,   // [512,4096]
                                 const float* __restrict__ b_ih,    // [4096]
                                 const float* __restrict__ b_hh,    // [4096]
                                 float* __restrict__ c,             // [512,1024]
                                 unsigned short* __restrict__ hbf,  // [512,1024] bf16
                                 float* __restrict__ hp1,           // [512,1024]
                                 float* __restrict__ hp2,           // [512,1024]
                                 int t) {
    int idx = blockIdx.x * blockDim.x + threadIdx.x;   // 512*1024
    int b = idx / HH;
    int h = idx % HH;
    const float* g = gates + (size_t)b * GG;

    float ig = g[h]          + b_ih[h]          + b_hh[h];
    float fg = g[HH + h]     + b_ih[HH + h]     + b_hh[HH + h];
    float gg = g[2 * HH + h] + b_ih[2 * HH + h] + b_hh[2 * HH + h];
    float og = g[3 * HH + h] + b_ih[3 * HH + h] + b_hh[3 * HH + h];

    float c_new = sigmoidf_fast(fg) * c[idx] + sigmoidf_fast(ig) * tanhf(gg);
    float h_new = sigmoidf_fast(og) * tanhf(c_new);

    c[idx]   = c_new;
    hbf[idx] = f32_to_bf16_rne(h_new);
    if (t == P1) hp1[idx] = h_new;
    if (t == P2) hp2[idx] = h_new;
}

// -------------------------------------------------------------------------
// Final FC (C=1) + sigmoid: out[2,512]; one wave32 per output row
// -------------------------------------------------------------------------
__global__ void fc_kernel(const float* __restrict__ hp1,
                          const float* __restrict__ hp2,
                          const float* __restrict__ w_fc,   // [1,1024]
                          const float* __restrict__ b_fc,   // [1]
                          float* __restrict__ out) {        // [2*512]
    int lane = threadIdx.x & 31;
    int wave = threadIdx.x >> 5;
    int row  = blockIdx.x * 8 + wave;                 // 0..1023
    const float* hv = (row < TB) ? (hp1 + (size_t)row * HH)
                                 : (hp2 + (size_t)(row - TB) * HH);
    float s = 0.0f;
    for (int k = lane; k < HH; k += 32)
        s += hv[k] * w_fc[k];
#pragma unroll
    for (int m = 16; m >= 1; m >>= 1)
        s += __shfl_xor(s, m, 32);
    if (lane == 0)
        out[row] = sigmoidf_fast(s + b_fc[0]);
}

// -------------------------------------------------------------------------
// Launch
// -------------------------------------------------------------------------
extern "C" void kernel_launch(void* const* d_in, const int* in_sizes, int n_in,
                              void* d_out, int out_size, void* d_ws, size_t ws_size,
                              hipStream_t stream) {
    (void)in_sizes; (void)n_in; (void)out_size; (void)ws_size;

    const float* x    = (const float*)d_in[0];   // [128,512,512]
    const float* w_ih = (const float*)d_in[1];   // [4096,512]
    const float* w_hh = (const float*)d_in[2];   // [4096,1024]
    const float* b_ih = (const float*)d_in[3];   // [4096]
    const float* b_hh = (const float*)d_in[4];   // [4096]
    const float* w_fc = (const float*)d_in[5];   // [1,1024]
    const float* b_fc = (const float*)d_in[6];   // [1]
    float* out = (float*)d_out;                  // [2,512]

    char* ws = (char*)d_ws;
    size_t off = 0;
    auto carve = [&](size_t bytes) -> char* {
        char* p = ws + off;
        off += (bytes + 255) & ~(size_t)255;
        return p;
    };
    unsigned short* wcat = (unsigned short*)carve((size_t)GG * KK * 2);          // 12.6 MB
    unsigned short* xbf  = (unsigned short*)carve((size_t)TSTEPS * TB * II * 2); // 18.9 MB
    unsigned short* hbf  = (unsigned short*)carve((size_t)TB * HH * 2);          //  1.0 MB
    float*          cbuf = (float*)carve((size_t)TB * HH * 4);                   //  2.1 MB
    float*          gts  = (float*)carve((size_t)TB * GG * 4);                   //  8.4 MB
    float*          hp1  = (float*)carve((size_t)TB * HH * 4);                   //  2.1 MB
    float*          hp2  = (float*)carve((size_t)TB * HH * 4);                   //  2.1 MB

    build_wcat_kernel<<<(GG * KK) / 256, 256, 0, stream>>>(w_ih, w_hh, wcat);
    convert_x_kernel<<<(TSTEPS * TB * II) / 256, 256, 0, stream>>>(x, xbf);
    init_state_kernel<<<(TB * HH) / 256, 256, 0, stream>>>(hbf, cbuf);

    dim3 ggrid(GG / 256, TB / 64);   // (16, 8)
    for (int t = 0; t < TSTEPS; ++t) {
        lstm_gemm_kernel<<<ggrid, 256, 0, stream>>>(xbf, hbf, wcat, gts, t);
        lstm_cell_kernel<<<(TB * HH) / 256, 256, 0, stream>>>(gts, b_ih, b_hh,
                                                              cbuf, hbf, hp1, hp2, t);
    }

    fc_kernel<<<(2 * TB) / 8, 256, 0, stream>>>(hp1, hp2, w_fc, b_fc, out);
}
